// Outer_Simplicial_2finder_74259984548101
// MI455X (gfx1250) — compile-verified, matrix operand order checked by reading, not run on hardware
//
#include <hip/hip_runtime.h>
#include <hip/hip_bf16.h>

// N=256, E=32768, H=2560.  Dominant work: 4 per-edge MLPs = ~378 GFLOP of GEMM.
// Strategy: everything through v_wmma_f32_16x16x32_bf16; fuse the 256->2560->256
// hidden pair per edge-tile so the [E,2560] activation never leaves the WGP.

typedef unsigned short u16;
typedef __attribute__((ext_vector_type(16))) __bf16 bf16x16;
typedef __attribute__((ext_vector_type(8)))  float  v8f;

#define NDIM 256
#define HDIM 2560
#define EDGES 32768
#define LDA 264   // padded LDS row stride (ushorts) for 256-wide tiles
#define LDH 136   // padded LDS row stride for 128-wide hidden chunk

__device__ __forceinline__ u16 f2bf(float f) {
  unsigned u = __builtin_bit_cast(unsigned, f);
  unsigned r = u + 0x7FFFu + ((u >> 16) & 1u);   // round-to-nearest-even
  return (u16)(r >> 16);
}
__device__ __forceinline__ float bf2f(u16 h) {
  unsigned v = ((unsigned)h) << 16;
  return __builtin_bit_cast(float, v);
}

union FragU { bf16x16 v; uint4 q[2]; };

// Fragment loader matching the ISA 16-bit A 16x32 layout (lanes 0-15: K blocks
// [8h..8h+7] and [16+8h..16+8h+7], h = lane>>4).  Works for A (row-major [M,K])
// and B (weights stored [Nout,K] row-major => B^T columns are contiguous rows).
__device__ __forceinline__ bf16x16 load_frag(const u16* base, int ld, int row0, int k0) {
  int lane = threadIdx.x & 31;
  const u16* p = base + (size_t)(row0 + (lane & 15)) * ld + k0 + ((lane >> 4) << 3);
  FragU f;
  f.q[0] = *(const uint4*)(p);
  f.q[1] = *(const uint4*)(p + 16);
  return f.v;
}

__device__ __forceinline__ v8f vzero() {
  v8f z = {0.f, 0.f, 0.f, 0.f, 0.f, 0.f, 0.f, 0.f};
  return z;
}

__device__ __forceinline__ v8f wmma_bf(bf16x16 a, bf16x16 b, v8f c) {
  return __builtin_amdgcn_wmma_f32_16x16x32_bf16(false, a, false, b, (short)0, c, false, false);
}

// C/D 16x16 f32 layout: vgpr v, lane l -> m = 8*(l>>4) + v, n = l&15.
__device__ __forceinline__ void store_tile_lds(u16* lds, int ld, int m0, int n0,
                                               v8f acc, const float* bias, bool relu) {
  int lane = threadIdx.x & 31;
  int nl = lane & 15, half = lane >> 4;
  float bv = bias[n0 + nl];
#pragma unroll
  for (int v = 0; v < 8; ++v) {
    float f = acc[v] + bv;
    if (relu) f = fmaxf(f, 0.f);
    lds[(m0 + half * 8 + v) * ld + n0 + nl] = f2bf(f);
  }
}

// ---------------------------------------------------------------------------
// Fused per-edge message MLP: a0 = x[i]*x[j];  a1 = relu(a0 W1^T + b1);
// a3 = relu( sum_chunks relu(a1 W2c^T + b2c) W3c^T + b3 );  m = a3 W4^T + b4;
// atomicAdd into agg[agg_idx].   64 edges per block, 8 waves, all WMMA bf16.
// ---------------------------------------------------------------------------
__global__ __launch_bounds__(256)
void msg_kernel(const int* __restrict__ iidx, const int* __restrict__ jidx,
                const int* __restrict__ aidx, const float* __restrict__ X,
                const u16* __restrict__ W1, const float* __restrict__ b1,
                const u16* __restrict__ W2, const float* __restrict__ b2,
                const u16* __restrict__ W3, const float* __restrict__ b3,
                const u16* __restrict__ W4, const float* __restrict__ b4,
                float* __restrict__ agg) {
  __shared__ u16 sA[64 * LDA];   // a0, then reused as hidden chunk + a3
  __shared__ u16 sB[64 * LDA];   // a1 (live through the chunk loop)
  u16* sH = sA;                  // hidden chunk aliases sA (a0 dead after layer1)

  const int tid  = threadIdx.x;
  const int e0   = blockIdx.x * 64;
  const int w    = tid >> 5;
  const int lane = tid & 31;
  const int mt   = (w & 3) * 16;        // 16-row tile base within the 64 edges
  const int cg   = (w >> 2) * 128;      // col group for 256-wide outputs (8 tiles)
  const int cg2  = (w >> 2) * 64;       // col group for 128-wide chunk (4 tiles)

  // stage 0: gather + elementwise product -> bf16 in sA
  for (int q = tid; q < 64 * 64; q += 256) {
    int r = q >> 6, c4 = q & 63;
    float4 a = ((const float4*)(X + (size_t)iidx[e0 + r] * NDIM))[c4];
    float4 b = ((const float4*)(X + (size_t)jidx[e0 + r] * NDIM))[c4];
    u16* d = sA + r * LDA + c4 * 4;
    d[0] = f2bf(a.x * b.x); d[1] = f2bf(a.y * b.y);
    d[2] = f2bf(a.z * b.z); d[3] = f2bf(a.w * b.w);
  }
  __syncthreads();

  // layer 1: sB = relu(sA @ W1^T + b1)   [64,256], K=256
  {
    v8f acc[8];
#pragma unroll
    for (int c = 0; c < 8; ++c) acc[c] = vzero();
#pragma unroll
    for (int ks = 0; ks < 8; ++ks) {
      bf16x16 fa = load_frag(sA, LDA, mt, ks * 32);
#pragma unroll
      for (int c = 0; c < 8; ++c) {
        bf16x16 fb = load_frag(W1, NDIM, cg + c * 16, ks * 32);
        acc[c] = wmma_bf(fa, fb, acc[c]);
      }
    }
#pragma unroll
    for (int c = 0; c < 8; ++c)
      store_tile_lds(sB, LDA, mt, cg + c * 16, acc[c], b1, true);
  }
  __syncthreads();

  // layers 2+3 fused over H=2560 in chunks of 128 (a2 never hits memory)
  v8f acc3[8];
#pragma unroll
  for (int c = 0; c < 8; ++c) acc3[c] = vzero();

  for (int hc = 0; hc < HDIM / 128; ++hc) {
    v8f acc2[4];
#pragma unroll
    for (int c = 0; c < 4; ++c) acc2[c] = vzero();
#pragma unroll
    for (int ks = 0; ks < 8; ++ks) {
      bf16x16 fa = load_frag(sB, LDA, mt, ks * 32);
#pragma unroll
      for (int c = 0; c < 4; ++c) {
        bf16x16 fb = load_frag(W2, NDIM, hc * 128 + cg2 + c * 16, ks * 32);
        acc2[c] = wmma_bf(fa, fb, acc2[c]);
      }
    }
#pragma unroll
    for (int c = 0; c < 4; ++c)
      store_tile_lds(sH, LDH, mt, cg2 + c * 16, acc2[c], b2 + hc * 128, true);
    __syncthreads();

#pragma unroll
    for (int ks = 0; ks < 4; ++ks) {
      bf16x16 fa = load_frag(sH, LDH, mt, ks * 32);
#pragma unroll
      for (int c = 0; c < 8; ++c) {
        bf16x16 fb = load_frag(W3, HDIM, cg + c * 16, hc * 128 + ks * 32);
        acc3[c] = wmma_bf(fa, fb, acc3[c]);
      }
    }
    __syncthreads();
  }

  // a3 = relu(acc3 + b3) -> sA
#pragma unroll
  for (int c = 0; c < 8; ++c)
    store_tile_lds(sA, LDA, mt, cg + c * 16, acc3[c], b3, true);
  __syncthreads();

  // layer 4 + scatter-add into agg
  {
    v8f acc[8];
#pragma unroll
    for (int c = 0; c < 8; ++c) acc[c] = vzero();
#pragma unroll
    for (int ks = 0; ks < 8; ++ks) {
      bf16x16 fa = load_frag(sA, LDA, mt, ks * 32);
#pragma unroll
      for (int c = 0; c < 8; ++c) {
        bf16x16 fb = load_frag(W4, NDIM, cg + c * 16, ks * 32);
        acc[c] = wmma_bf(fa, fb, acc[c]);
      }
    }
    int nl = lane & 15, half = lane >> 4;
#pragma unroll
    for (int c = 0; c < 8; ++c) {
      int n = cg + c * 16 + nl;
      float bv = b4[n];
#pragma unroll
      for (int v = 0; v < 8; ++v) {
        int r = mt + half * 8 + v;
        int row = aidx[e0 + r];
        atomicAdd(agg + (size_t)row * NDIM + n, acc[c][v] + bv);
      }
    }
  }
}

// ---------------------------------------------------------------------------
// Generic bf16 WMMA GEMM:  C[m,n] = act(A[m,:] . W[n,:] + bias[n])
// A: [M,K] bf16 row-major, W: [N,K] bf16 row-major.  64x256 tile per block.
// ---------------------------------------------------------------------------
__global__ __launch_bounds__(256)
void gemm_kernel(const u16* __restrict__ A, const u16* __restrict__ W,
                 const float* __restrict__ bias, float* __restrict__ Cf, int ldcf,
                 u16* __restrict__ Cb, int N, int K, int relu) {
  const int tid = threadIdx.x;
  const int w = tid >> 5, lane = tid & 31;
  const int m0 = blockIdx.x * 64 + (w & 3) * 16;
  const int n0 = blockIdx.y * 256 + (w >> 2) * 128;

  v8f acc[8];
#pragma unroll
  for (int c = 0; c < 8; ++c) acc[c] = vzero();

#pragma unroll 1
  for (int ks = 0; ks < K; ks += 32) {
    bf16x16 fa = load_frag(A, K, m0, ks);
#pragma unroll
    for (int c = 0; c < 8; ++c) {
      bf16x16 fb = load_frag(W, K, n0 + c * 16, ks);
      acc[c] = wmma_bf(fa, fb, acc[c]);
    }
  }

  int nl = lane & 15, half = lane >> 4;
#pragma unroll
  for (int c = 0; c < 8; ++c) {
    int n = n0 + c * 16 + nl;
    float bv = bias[n];
#pragma unroll
    for (int v = 0; v < 8; ++v) {
      float f = acc[c][v] + bv;
      if (relu) f = fmaxf(f, 0.f);
      int m = m0 + half * 8 + v;
      if (Cb) Cb[(size_t)m * N + n] = f2bf(f);
      if (Cf) Cf[(size_t)m * ldcf + n] = f;
    }
  }
}

// ------------------------------- small kernels -----------------------------
__global__ void transpose_kernel(const float* __restrict__ x, float* __restrict__ xT) {
  int i = blockIdx.x * 256 + threadIdx.x;      // 65536 elements
  int r = i >> 8, c = i & 255;
  xT[c * 256 + r] = x[i];
}

__global__ void f2bf_kernel(const float* __restrict__ s, u16* __restrict__ d, int n) {
  int i = blockIdx.x * 256 + threadIdx.x;
  if (i < n) d[i] = f2bf(s[i]);
}

__global__ void zero_kernel(float* __restrict__ p, int n) {
  int i = blockIdx.x * 256 + threadIdx.x;
  if (i < n) p[i] = 0.f;
}

__global__ void final_dot_kernel(const u16* __restrict__ u2, const u16* __restrict__ w3,
                                 const float* __restrict__ b3, float* __restrict__ out) {
  int r = threadIdx.x;                          // 256 rows, one per thread
  float s = b3[0];
  for (int c = 0; c < 256; ++c) s += bf2f(u2[r * 256 + c]) * bf2f(w3[c]);
  out[r] = s;
}

// ---------------------------------------------------------------------------
extern "C" void kernel_launch(void* const* d_in, const int* in_sizes, int n_in,
                              void* d_out, int out_size, void* d_ws, size_t ws_size,
                              hipStream_t stream) {
  // d_in layout (pytree order): 0 edge_index[2,E], 1 x[256,256],
  // fwd_targets W/b x4 (2..9), fwd_sources (10..17), fwd_mlp2 (18..23),
  // bwd_targets (24..31), bwd_sources (32..39), bwd_mlp2 (40..45), mlp_2s (46..51)
  const int* ei = (const int*)d_in[0];
  const float* x = (const float*)d_in[1];
  char* ws = (char*)d_ws;
  (void)n_in; (void)out_size; (void)ws_size;

  auto al = [](size_t v) { return (v + 255) & ~(size_t)255; };

  size_t off = 0;
  float* xT = (float*)(ws + off); off += al(65536 * sizeof(float));

  static const int wmi[25] = {2, 4, 6, 8, 10, 12, 14, 16, 18, 20, 22,
                              24, 26, 28, 30, 32, 34, 36, 38, 40, 42, 44,
                              46, 48, 50};
  size_t woff[52];
  for (int t = 0; t < 25; ++t) { woff[wmi[t]] = off; off += al((size_t)in_sizes[wmi[t]] * 2); }

  float* agg      = (float*)(ws + off); off += al(4 * 65536 * sizeof(float));
  float* concat   = (float*)(ws + off); off += al(256 * 1024 * sizeof(float));
  u16* concat_bf  = (u16*)(ws + off);   off += al((size_t)256 * 1024 * 2);
  u16* agg_bf     = (u16*)(ws + off);   off += al((size_t)65536 * 2);
  u16* t1_bf      = (u16*)(ws + off);   off += al((size_t)256 * 2560 * 2);
  u16* t2_bf      = (u16*)(ws + off);   off += al((size_t)65536 * 2);

  auto WB = [&](int i) { return (const u16*)(ws + woff[i]); };
  auto FP = [&](int i) { return (const float*)d_in[i]; };

  // prep: transpose x, convert weights to bf16, zero accumulators
  hipLaunchKernelGGL(transpose_kernel, dim3(256), dim3(256), 0, stream, x, xT);
  for (int t = 0; t < 25; ++t) {
    int n = in_sizes[wmi[t]];
    hipLaunchKernelGGL(f2bf_kernel, dim3((n + 255) / 256), dim3(256), 0, stream,
                       (const float*)d_in[wmi[t]], (u16*)(ws + woff[wmi[t]]), n);
  }
  hipLaunchKernelGGL(zero_kernel, dim3(1024), dim3(256), 0, stream, agg, 4 * 65536);

  const int* src = ei;
  const int* dst = ei + EDGES;

  // four fused message passes (o1..o4 aggregation buffers)
  hipLaunchKernelGGL(msg_kernel, dim3(EDGES / 64), dim3(256), 0, stream,
      dst, src, dst, x,  WB(2),  FP(3),  WB(4),  FP(5),  WB(6),  FP(7),  WB(8),  FP(9),  agg + 0 * 65536);
  hipLaunchKernelGGL(msg_kernel, dim3(EDGES / 64), dim3(256), 0, stream,
      dst, src, dst, xT, WB(10), FP(11), WB(12), FP(13), WB(14), FP(15), WB(16), FP(17), agg + 1 * 65536);
  hipLaunchKernelGGL(msg_kernel, dim3(EDGES / 64), dim3(256), 0, stream,
      src, dst, src, x,  WB(24), FP(25), WB(26), FP(27), WB(28), FP(29), WB(30), FP(31), agg + 2 * 65536);
  hipLaunchKernelGGL(msg_kernel, dim3(EDGES / 64), dim3(256), 0, stream,
      src, dst, src, xT, WB(32), FP(33), WB(34), FP(35), WB(36), FP(37), WB(38), FP(39), agg + 3 * 65536);

  // update MLPs (fwd_mlp2 shared by o1/o2, bwd_mlp2 by o3/o4) -> concat cols
  for (int k = 0; k < 4; ++k) {
    int p = (k < 2) ? 18 : 40;
    hipLaunchKernelGGL(f2bf_kernel, dim3(256), dim3(256), 0, stream,
                       agg + k * 65536, agg_bf, 65536);
    hipLaunchKernelGGL(gemm_kernel, dim3(4, 10), dim3(256), 0, stream,
        agg_bf, WB(p), FP(p + 1), (float*)nullptr, 0, t1_bf, HDIM, NDIM, 1);
    hipLaunchKernelGGL(gemm_kernel, dim3(4, 1), dim3(256), 0, stream,
        t1_bf, WB(p + 2), FP(p + 3), (float*)nullptr, 0, t2_bf, NDIM, HDIM, 1);
    hipLaunchKernelGGL(gemm_kernel, dim3(4, 1), dim3(256), 0, stream,
        t2_bf, WB(p + 4), FP(p + 5), concat + k * 256, 1024, (u16*)nullptr, NDIM, NDIM, 0);
  }

  // final MLP: [256,1024] -> 2560 -> 256 -> 1
  hipLaunchKernelGGL(f2bf_kernel, dim3(1024), dim3(256), 0, stream, concat, concat_bf, 262144);
  hipLaunchKernelGGL(gemm_kernel, dim3(4, 10), dim3(256), 0, stream,
      concat_bf, WB(46), FP(47), (float*)nullptr, 0, t1_bf, HDIM, 1024, 1);
  hipLaunchKernelGGL(gemm_kernel, dim3(4, 1), dim3(256), 0, stream,
      t1_bf, WB(48), FP(49), (float*)nullptr, 0, t2_bf, NDIM, HDIM, 1);
  hipLaunchKernelGGL(final_dot_kernel, dim3(1), dim3(256), 0, stream,
      t2_bf, WB(50), FP(51), (float*)d_out);
}
